// GATEncoder_34600256537270
// MI455X (gfx1250) — matrix operand order repaired and checked by reading, not compiled, not run on hardware
//
#include <hip/hip_runtime.h>
#include <hip/hip_bf16.h>
#include <math.h>

typedef __attribute__((ext_vector_type(16))) _Float16 v16h;
typedef __attribute__((ext_vector_type(8)))  _Float16 v8h;
typedef __attribute__((ext_vector_type(8)))  float    v8f;

#define F_IN   256
#define F_OUT  128   // H*C
#define HEADS  4
#define CPH    32
#define NEG_SLOPE 0.2f
#define KEEP_SCALE 1.25f   // 1/(1-0.2)

// ---------- deterministic dropout mask (hash of element index) ----------
__device__ __forceinline__ float dmask(unsigned idx) {
    unsigned s = idx * 2654435761u;
    s ^= s >> 15; s *= 0x2c1b3c6du;
    s ^= s >> 12; s *= 0x297a2d39u;
    s ^= s >> 15;
    // keep with prob 0.8:  0.8 * 2^24 = 13421772.8
    return ((s & 0x00FFFFFFu) < 13421773u) ? KEEP_SCALE : 0.0f;
}

__device__ __forceinline__ void atomicMaxF(float* addr, float val) {
    int old = __float_as_int(*addr);
    while (__int_as_float(old) < val) {
        int assumed = old;
        old = atomicCAS((int*)addr, assumed, __float_as_int(val));
        if (old == assumed) break;
    }
}

// ---------------- K0: init out / m / den ----------------
__global__ void gat_init(float* __restrict__ out, float* __restrict__ m,
                         float* __restrict__ den, int out_n, int nh) {
    int i = blockIdx.x * blockDim.x + threadIdx.x;
    if (i < out_n) out[i] = 0.0f;
    if (i < nh) {
        m[i]   = -INFINITY;
        den[i] = 0.0f;
    }
}

// ---------------- K1: dropout(x) @ W -> h, via WMMA f16 ----------------
// Block: 256 threads = 8 waves; block handles 128 rows; each wave one 16-row
// M tile across all 8 N tiles (128 cols).  W staged to LDS as f16 transposed
// (Wt[n][k_local], stride 136 halves = 272B, 16B aligned) in two K-halves.
__global__ __launch_bounds__(256) void gat_gemm(const float* __restrict__ x,
                                                const float* __restrict__ W,
                                                float* __restrict__ h,
                                                int nNodes) {
    __shared__ __align__(16) _Float16 Wt[128 * 136];   // 34,816 B

    const int t    = threadIdx.x;
    const int wave = t >> 5;
    const int lane = t & 31;
    const int l    = lane & 15;   // column / row-in-tile lane
    const int g    = lane >> 4;   // K-group select

    const int mBase = blockIdx.x * 128 + wave * 16;
    const int row   = mBase + l;          // A-matrix row for this lane
    const bool rv   = row < nNodes;
    const float* xr = x + (long)row * F_IN;
    const unsigned rbase = (unsigned)row * (unsigned)F_IN;

    v8f acc[8];
    #pragma unroll
    for (int i = 0; i < 8; ++i) acc[i] = (v8f)0.0f;

    #pragma unroll
    for (int half = 0; half < 2; ++half) {
        __syncthreads();
        // stage K rows [half*128, half*128+128) of W, transposed, f16
        for (int i = t; i < 128 * 128; i += 256) {
            int kl = i >> 7;          // local k
            int n  = i & 127;         // output column
            Wt[n * 136 + kl] = (_Float16)W[(half * 128 + kl) * F_OUT + n];
        }
        __syncthreads();

        #pragma unroll
        for (int kk = 0; kk < 4; ++kk) {
            const int k0g = half * 128 + kk * 32;  // global K base (x)
            const int k0l = kk * 32;               // local K base (LDS)

            // ---- A fragment: 16x32 f16, ISA layout ----
            // lane group g=0: K {k0..k0+7, k0+16..k0+23}
            // lane group g=1: K {k0+8..k0+15, k0+24..k0+31}
            v16h afrag;
            if (rv) {
                const int ka = k0g + g * 8;
                const int kb = ka + 16;
                float4 a0 = *(const float4*)(xr + ka);
                float4 a1 = *(const float4*)(xr + ka + 4);
                float4 b0 = *(const float4*)(xr + kb);
                float4 b1 = *(const float4*)(xr + kb + 4);
                float va[16] = {a0.x, a0.y, a0.z, a0.w, a1.x, a1.y, a1.z, a1.w,
                                b0.x, b0.y, b0.z, b0.w, b1.x, b1.y, b1.z, b1.w};
                #pragma unroll
                for (int e = 0; e < 16; ++e) {
                    int kg = (e < 8) ? (ka + e) : (kb + e - 8);
                    afrag[e] = (_Float16)(va[e] * dmask(rbase + (unsigned)kg));
                }
            } else {
                #pragma unroll
                for (int e = 0; e < 16; ++e) afrag[e] = (_Float16)0.0f;
            }

            // ---- 8 N tiles: B fragment from LDS, WMMA ----
            #pragma unroll
            for (int tn = 0; tn < 8; ++tn) {
                // B 32x16 f16 layout: lane (g,l): column n=tile*16+l,
                // elements e=0..7 hold K = k0 + g*8 + e, e=8..15 hold K = k0 + 16 + g*8 + (e-8)
                const _Float16* bp = &Wt[(tn * 16 + l) * 136 + k0l + g * 8];
                union { v16h v; v8h p[2]; } bu;
                bu.p[0] = *(const v8h*)bp;
                bu.p[1] = *(const v8h*)(bp + 16);
                acc[tn] = __builtin_amdgcn_wmma_f32_16x16x32_f16(
                    false, afrag, false, bu.v, (short)0, acc[tn], false, false);
            }
        }
    }

    // ---- store h: C/D layout: VGPR r -> row (r + 8*g), col tn*16 + l ----
    #pragma unroll
    for (int r = 0; r < 8; ++r) {
        int grow = mBase + r + 8 * g;
        if (grow < nNodes) {
            float* hp = h + (long)grow * F_OUT + l;
            #pragma unroll
            for (int tn = 0; tn < 8; ++tn) hp[tn * 16] = acc[tn][r];
        }
    }
}

// ---------------- K2: per-node attention logits ----------------
__global__ void gat_att(const float* __restrict__ h,
                        const float* __restrict__ att_src,
                        const float* __restrict__ att_dst,
                        float* __restrict__ a_s, float* __restrict__ a_d,
                        int nNodes) {
    int id = blockIdx.x * blockDim.x + threadIdx.x;
    int n  = id >> 2;
    int hd = id & 3;
    if (n >= nNodes) return;
    const float* hp = h + (long)n * F_OUT + hd * CPH;
    const float* as = att_src + hd * CPH;
    const float* ad = att_dst + hd * CPH;
    float s = 0.0f, d = 0.0f;
    #pragma unroll
    for (int c = 0; c < CPH; ++c) {
        float v = hp[c];
        s += v * as[c];
        d += v * ad[c];
    }
    a_s[n * HEADS + hd] = s;
    a_d[n * HEADS + hd] = d;
}

// ---------------- K3: edge pass 1: e = lrelu(a_s[src]+a_d[dst]); segmax ----
__global__ void gat_edge1(const long long* __restrict__ eidx,
                          const float* __restrict__ a_s,
                          const float* __restrict__ a_d,
                          float* __restrict__ ebuf, float* __restrict__ m,
                          int E, int nNodes) {
    int id = blockIdx.x * blockDim.x + threadIdx.x;
    int tot = (E + nNodes) * HEADS;
    if (id >= tot) return;
    int ei = id >> 2;
    int hd = id & 3;
    int s, d;
    if (ei < E) { s = (int)eidx[ei]; d = (int)eidx[E + ei]; }
    else        { s = d = ei - E; }
    float v = a_s[s * HEADS + hd] + a_d[d * HEADS + hd];
    v = (v < 0.0f) ? NEG_SLOPE * v : v;
    ebuf[id] = v;
    atomicMaxF(&m[d * HEADS + hd], v);
}

// ---------------- K4: edge pass 2: ee = exp(e - m[dst]); segsum ----------
__global__ void gat_edge2(const long long* __restrict__ eidx,
                          float* __restrict__ ebuf,
                          const float* __restrict__ m, float* __restrict__ den,
                          int E, int nNodes) {
    int id = blockIdx.x * blockDim.x + threadIdx.x;
    int tot = (E + nNodes) * HEADS;
    if (id >= tot) return;
    int ei = id >> 2;
    int hd = id & 3;
    int d = (ei < E) ? (int)eidx[E + ei] : (ei - E);
    float ee = __expf(ebuf[id] - m[d * HEADS + hd]);
    ebuf[id] = ee;
    atomicAdd(&den[d * HEADS + hd], ee);
}

// ---------------- K5: edge pass 3: out[dst] += alpha * h[src] ------------
__global__ void gat_edge3(const long long* __restrict__ eidx,
                          const float* __restrict__ ebuf,
                          const float* __restrict__ den,
                          const float* __restrict__ h, float* __restrict__ out,
                          int E, int nNodes) {
    int id = blockIdx.x * blockDim.x + threadIdx.x;
    int tot = (E + nNodes) * F_OUT;
    if (id >= tot) return;
    int ei = id >> 7;
    int c  = id & 127;
    int hd = c >> 5;
    int s, d;
    if (ei < E) { s = (int)eidx[ei]; d = (int)eidx[E + ei]; }
    else        { s = d = ei - E; }
    float alpha = ebuf[ei * HEADS + hd] /
                  fmaxf(den[d * HEADS + hd], 1e-16f);
    atomicAdd(&out[d * F_OUT + c], alpha * h[(long)s * F_OUT + c]);
}

// ---------------- K6: bias + PReLU ----------------
__global__ void gat_final(float* __restrict__ out, const float* __restrict__ bias,
                          const float* __restrict__ prelu_a, int out_n) {
    int i = blockIdx.x * blockDim.x + threadIdx.x;
    if (i >= out_n) return;
    float v = out[i] + bias[i & 127];
    float a = prelu_a[0];
    out[i] = (v >= 0.0f) ? v : a * v;
}

extern "C" void kernel_launch(void* const* d_in, const int* in_sizes, int n_in,
                              void* d_out, int out_size, void* d_ws, size_t ws_size,
                              hipStream_t stream) {
    const float*     x        = (const float*)d_in[0];
    const float*     W        = (const float*)d_in[1];
    const float*     att_src  = (const float*)d_in[2];
    const float*     att_dst  = (const float*)d_in[3];
    const float*     bias     = (const float*)d_in[4];
    const float*     prelu_a  = (const float*)d_in[5];
    const long long* eidx     = (const long long*)d_in[6];

    const int Nn = in_sizes[0] / F_IN;     // 50000
    const int E  = in_sizes[6] / 2;        // 1600000
    float* out = (float*)d_out;

    // workspace layout (floats)
    float* h    = (float*)d_ws;                   // Nn*128
    float* a_s  = h    + (size_t)Nn * F_OUT;      // Nn*4
    float* a_d  = a_s  + (size_t)Nn * HEADS;      // Nn*4
    float* m    = a_d  + (size_t)Nn * HEADS;      // Nn*4
    float* den  = m    + (size_t)Nn * HEADS;      // Nn*4
    float* ebuf = den  + (size_t)Nn * HEADS;      // (E+Nn)*4

    const int threads = 256;
    const int nh      = Nn * HEADS;
    const int totE    = (E + Nn) * HEADS;
    const int totC    = (E + Nn) * F_OUT;

    gat_init<<<(out_size + threads - 1) / threads, threads, 0, stream>>>(
        out, m, den, out_size, nh);

    gat_gemm<<<(Nn + 127) / 128, threads, 0, stream>>>(x, W, h, Nn);

    gat_att<<<(nh + threads - 1) / threads, threads, 0, stream>>>(
        h, att_src, att_dst, a_s, a_d, Nn);

    gat_edge1<<<(totE + threads - 1) / threads, threads, 0, stream>>>(
        eidx, a_s, a_d, ebuf, m, E, Nn);

    gat_edge2<<<(totE + threads - 1) / threads, threads, 0, stream>>>(
        eidx, ebuf, m, den, E, Nn);

    gat_edge3<<<(totC + threads - 1) / threads, threads, 0, stream>>>(
        eidx, ebuf, den, h, out, E, Nn);

    gat_final<<<(out_size + threads - 1) / threads, threads, 0, stream>>>(
        out, bias, prelu_a, out_size);
}
